// Point_Embedding_17145509446499
// MI455X (gfx1250) — matrix operand order repaired and checked by reading, not compiled, not run on hardware
//
#include <hip/hip_runtime.h>
#include <hip/hip_fp16.h>

typedef __attribute__((ext_vector_type(16))) _Float16 v16h;
typedef __attribute__((ext_vector_type(8)))  float    v8f;

#define B_   8
#define N_   2048
#define K_   32
#define CO_  64
#define NEG_SLOPE_ 0.2f

// ---------------- WMMA fragment loaders (16-bit layouts per CDNA5 ISA 7.12.2) ----
union Frag16 { v16h v; uint4 q[2]; };

// A-matrix 16x32 f16: lanes0-15 (M=lane): halves K[0..8)+K[16..24); lanes16-31: K[8..16)+K[24..32)
__device__ __forceinline__ v16h load_a_frag(const _Float16* rowbase, int lane) {
  int hi = (lane >> 4) & 1;
  const uint4* p = reinterpret_cast<const uint4*>(rowbase + hi * 8);
  Frag16 f; f.q[0] = p[0]; f.q[1] = p[2];
  return f.v;
}
// B-matrix 32x16 f16: col = lane&15; lanes0-15: K[0..8)+K[16..24); lanes16-31: K[8..16)+K[24..32)
__device__ __forceinline__ v16h load_b_frag(const _Float16* colbase, int lane) {
  int hi = (lane >> 4) & 1;
  const uint4* p = reinterpret_cast<const uint4*>(colbase + hi * 8);
  Frag16 f; f.q[0] = p[0]; f.q[1] = p[2];
  return f.v;
}

// ---------------- register-resident top-32 (replace-worst, fully unrolled) -------
struct TopK {
  float d[K_]; int id[K_]; float worst; int wpos;
  __device__ __forceinline__ void init() {
#pragma unroll
    for (int i = 0; i < K_; ++i) { d[i] = 3.4e38f; id[i] = 0; }
    worst = 3.4e38f; wpos = 0;
  }
  __device__ __forceinline__ void insert(float nd, int ni) {
    if (nd < worst) {
#pragma unroll
      for (int i = 0; i < K_; ++i) if (i == wpos) { d[i] = nd; id[i] = ni; }
      worst = d[0]; wpos = 0;
#pragma unroll
      for (int i = 1; i < K_; ++i) if (d[i] > worst) { worst = d[i]; wpos = i; }
    }
  }
};

// ---------------- kNN for block 1 (C=3, scalar; all points fit in LDS) -----------
__global__ __launch_bounds__(256) void knn_xyz(const float* __restrict__ a,
                                               int* __restrict__ idxout) {
  __shared__ float P[N_ * 3];
  const int b = blockIdx.x, tid = threadIdx.x;
  for (int e = tid; e < N_ * 3; e += 256) P[e] = a[b * N_ * 3 + e];
  __syncthreads();
  const int q = blockIdx.y * 256 + tid;
  const float qx = P[q * 3 + 0], qy = P[q * 3 + 1], qz = P[q * 3 + 2];
  TopK tk; tk.init();
  for (int c = 0; c < N_; ++c) {
    float dx = P[c * 3 + 0] - qx, dy = P[c * 3 + 1] - qy, dz = P[c * 3 + 2] - qz;
    tk.insert(dx * dx + dy * dy + dz * dz, c);
  }
#pragma unroll
  for (int i = 0; i < K_; ++i) idxout[(b * N_ + q) * K_ + i] = tk.id[i];
}

// ---------------- kNN for block 2 (C=64): WMMA Gram tiles + register top-k -------
__global__ __launch_bounds__(128) void knn_feat(const _Float16* __restrict__ hh,
                                                int* __restrict__ idxout) {
  __shared__ __align__(16) _Float16 Q[64 * 64];
  __shared__ __align__(16) _Float16 Cc[64 * 64];
  __shared__ float Dt[64 * 64];
  __shared__ float sqc[64];
  const int b = blockIdx.x, qbase = blockIdx.y * 64;
  const int tid = threadIdx.x, lane = tid & 31, w = tid >> 5;

  for (int e = tid; e < 64 * 64; e += 128) Q[e] = hh[((size_t)b * N_ + qbase) * CO_ + e];
  TopK tk; tk.init();

  for (int cc = 0; cc < N_ / 64; ++cc) {
    const int cbase = cc * 64;
    __syncthreads();                                  // Dt/Cc consumed; Q visible (iter 0)
    for (int e = tid; e < 64 * 64; e += 128) Cc[e] = hh[((size_t)b * N_ + cbase) * CO_ + e];
    if (cc + 1 < N_ / 64)
      __builtin_prefetch(&hh[((size_t)b * N_ + cbase + 64) * CO_], 0, 1);
    __syncthreads();
    if (tid < 64) {                                   // |c|^2 (query norm is rank-invariant)
      float s = 0.f;
#pragma unroll 8
      for (int i = 0; i < 64; ++i) { float v = (float)Cc[tid * 64 + i]; s += v * v; }
      sqc[tid] = s;
    }
    // Gram tile: wave w computes rows [16w,16w+16) x 64 candidate cols, K=64
    const int arow = w * 16 + (lane & 15);
#pragma unroll
    for (int nt = 0; nt < 4; ++nt) {
      const int bcol = nt * 16 + (lane & 15);
      v8f acc = {};
#pragma unroll
      for (int ks = 0; ks < 2; ++ks) {
        v16h af = load_a_frag(&Q[arow * 64 + ks * 32], lane);
        v16h bf = load_b_frag(&Cc[bcol * 64 + ks * 32], lane);
        acc = __builtin_amdgcn_wmma_f32_16x16x32_f16(false, af, false, bf,
                                                     (short)0, acc, false, false);
      }
      const int mb = w * 16 + ((lane >> 4) & 1) * 8;
      const int nc = nt * 16 + (lane & 15);
#pragma unroll
      for (int j = 0; j < 8; ++j) Dt[(mb + j) * 64 + nc] = acc[j];
    }
    __syncthreads();
    if (tid < 64) {
#pragma unroll 4
      for (int c = 0; c < 64; ++c)
        tk.insert(sqc[c] - 2.0f * Dt[tid * 64 + c], cbase + c);
    }
  }
  if (tid < 64) {
    const int q = qbase + tid;
#pragma unroll
    for (int i = 0; i < K_; ++i) idxout[(b * N_ + q) * K_ + i] = tk.id[i];
  }
}

// ---------------- EdgeConv GEMM pass -------------------------------------------
// pass A (FINAL=false): accumulate per-channel sum / sumsq of y (BN stats)
// pass B (FINAL=true) : recompute y, apply BN scale/shift + LeakyReLU, max over k
template <int CIN, int KPAD, bool FINAL, bool FIRST>
__global__ __launch_bounds__(128) void edgeconv_pass(
    const float* __restrict__ xf, const _Float16* __restrict__ hh,
    const int* __restrict__ idx, const float* __restrict__ W,
    float* __restrict__ sums, const float* __restrict__ ss,
    _Float16* __restrict__ outh, float* __restrict__ outf) {
  __shared__ __align__(16) _Float16 Wl[CO_ * KPAD];
  __shared__ __align__(16) _Float16 feat[4 * 32 * KPAD];
  __shared__ float wgS[CO_], wgS2[CO_];
  const int b = blockIdx.x, tid = threadIdx.x, lane = tid & 31, w = tid >> 5;

  for (int e = tid; e < CO_ * KPAD; e += 128) {        // W -> f16, zero-padded to KPAD
    int o = e / KPAD, kk = e % KPAD;
    Wl[e] = (kk < 2 * CIN) ? (_Float16)W[o * (2 * CIN) + kk] : (_Float16)0.f;
  }
  if (!FINAL && tid < CO_) { wgS[tid] = 0.f; wgS2[tid] = 0.f; }
  __syncthreads();

  _Float16* fw = &feat[w * 32 * KPAD];                  // wave-private feat region
  for (int p = 0; p < 16; ++p) {
    const int n = blockIdx.y * 64 + w * 16 + p;
    {                                                   // lane = neighbor slot kk
      const int nb = idx[(b * N_ + n) * K_ + lane];
      _Float16* fr = fw + lane * KPAD;
#pragma unroll 4
      for (int c = 0; c < CIN; ++c) {
        float ctr = FIRST ? xf[((size_t)b * N_ + n) * CIN + c]
                          : (float)hh[((size_t)b * N_ + n) * CO_ + c];
        float nbv = FIRST ? xf[((size_t)b * N_ + nb) * CIN + c]
                          : (float)hh[((size_t)b * N_ + nb) * CO_ + c];
        fr[c] = (_Float16)(nbv - ctr);
        fr[CIN + c] = (_Float16)ctr;
      }
      if (2 * CIN < KPAD) {
#pragma unroll
        for (int c = 2 * CIN; c < KPAD; ++c) fr[c] = (_Float16)0.f;
      }
    }
#pragma unroll
    for (int nt = 0; nt < 4; ++nt) {
      const int ch = nt * 16 + (lane & 15);
      float vmax = -3.4e38f, s = 0.f, s2 = 0.f;
#pragma unroll
      for (int mt = 0; mt < 2; ++mt) {
        v8f acc = {};
        const int arow = mt * 16 + (lane & 15);
#pragma unroll
        for (int ks = 0; ks < KPAD / 32; ++ks) {
          v16h af = load_a_frag(fw + arow * KPAD + ks * 32, lane);
          v16h bf = load_b_frag(&Wl[ch * KPAD + ks * 32], lane);
          acc = __builtin_amdgcn_wmma_f32_16x16x32_f16(false, af, false, bf,
                                                       (short)0, acc, false, false);
        }
        if (FINAL) {
          const float sc = ss[ch], sh = ss[CO_ + ch];
#pragma unroll
          for (int j = 0; j < 8; ++j) {
            float v = acc[j] * sc + sh;
            v = (v >= 0.f) ? v : NEG_SLOPE_ * v;
            vmax = fmaxf(vmax, v);
          }
        } else {
#pragma unroll
          for (int j = 0; j < 8; ++j) { s += acc[j]; s2 += acc[j] * acc[j]; }
        }
      }
      if (FINAL) {
        vmax = fmaxf(vmax, __shfl_xor(vmax, 16, 32));   // merge the two lane halves
        if (lane < 16) {
          if (FIRST) outh[((size_t)b * N_ + n) * CO_ + ch] = (_Float16)vmax;
          else       outf[((size_t)b * N_ + n) * CO_ + ch] = vmax;
        }
      } else {
        atomicAdd(&wgS[ch], s);
        atomicAdd(&wgS2[ch], s2);
      }
    }
  }
  if (!FINAL) {
    __syncthreads();
    if (tid < CO_) {
      atomicAdd(&sums[tid], wgS[tid]);
      atomicAdd(&sums[CO_ + tid], wgS2[tid]);
    }
  }
}

// ---------------- BN stat finalize + init ---------------------------------------
__global__ void zero_sums(float* s) { s[threadIdx.x] = 0.f; }

__global__ void bn_finalize(const float* __restrict__ sums, const float* __restrict__ gamma,
                            const float* __restrict__ beta, float* __restrict__ ss,
                            float invM) {
  const int c = threadIdx.x;
  const float mean = sums[c] * invM;
  const float var  = sums[CO_ + c] * invM - mean * mean;
  const float sc   = gamma[c] * rsqrtf(var + 1e-5f);
  ss[c] = sc;
  ss[CO_ + c] = beta[c] - mean * sc;
}

// ---------------- host side ------------------------------------------------------
extern "C" void kernel_launch(void* const* d_in, const int* in_sizes, int n_in,
                              void* d_out, int out_size, void* d_ws, size_t ws_size,
                              hipStream_t stream) {
  const float* a  = (const float*)d_in[0];
  // d_in[1] = k (== 32, hardcoded)
  const float* W1 = (const float*)d_in[2];
  const float* g1 = (const float*)d_in[3];
  const float* b1 = (const float*)d_in[4];
  const float* W2 = (const float*)d_in[5];
  const float* g2 = (const float*)d_in[6];
  const float* b2 = (const float*)d_in[7];

  char* ws = (char*)d_ws;
  int*       idx1  = (int*)ws;                                   // 2 MB
  int*       idx2  = (int*)(ws + (size_t)2 * 1024 * 1024);       // 2 MB
  _Float16*  hh    = (_Float16*)(ws + (size_t)4 * 1024 * 1024);  // 2 MB (h as f16)
  float*     sums1 = (float*)(ws + (size_t)6 * 1024 * 1024);
  float*     sums2 = sums1 + 2 * CO_;
  float*     ss1   = sums2 + 2 * CO_;
  float*     ss2   = ss1 + 2 * CO_;
  const float invM = 1.0f / (float)((size_t)B_ * N_ * K_);

  zero_sums<<<1, 256, 0, stream>>>(sums1);  // zeros sums1 + sums2 (contiguous 256 f32)

  knn_xyz<<<dim3(B_, N_ / 256), 256, 0, stream>>>(a, idx1);
  edgeconv_pass<3, 32, false, true><<<dim3(B_, N_ / 64), 128, 0, stream>>>(
      a, nullptr, idx1, W1, sums1, nullptr, nullptr, nullptr);
  bn_finalize<<<1, CO_, 0, stream>>>(sums1, g1, b1, ss1, invM);
  edgeconv_pass<3, 32, true, true><<<dim3(B_, N_ / 64), 128, 0, stream>>>(
      a, nullptr, idx1, W1, nullptr, ss1, hh, nullptr);

  knn_feat<<<dim3(B_, N_ / 64), 128, 0, stream>>>(hh, idx2);
  edgeconv_pass<64, 128, false, false><<<dim3(B_, N_ / 64), 128, 0, stream>>>(
      nullptr, hh, idx2, W2, sums2, nullptr, nullptr, nullptr);
  bn_finalize<<<1, CO_, 0, stream>>>(sums2, g2, b2, ss2, invM);
  edgeconv_pass<64, 128, true, false><<<dim3(B_, N_ / 64), 128, 0, stream>>>(
      nullptr, hh, idx2, W2, nullptr, ss2, nullptr, (float*)d_out);

  (void)in_sizes; (void)n_in; (void)out_size; (void)ws_size;
}
